// MultiHeadAttention_3590592660015
// MI455X (gfx1250) — matrix operand order repaired and checked by reading, not compiled, not run on hardware
//
#include <hip/hip_runtime.h>
#include <hip/hip_bf16.h>

// ---------- problem constants ----------
#define B_  2
#define S_  4096
#define D_  512
#define H_  8
#define DK_ 64

typedef _Float16 half8  __attribute__((ext_vector_type(8)));
typedef _Float16 half16 __attribute__((ext_vector_type(16)));
typedef float    float8 __attribute__((ext_vector_type(8)));
typedef int      int4v  __attribute__((ext_vector_type(4)));

typedef __attribute__((address_space(1))) int4v* g_int4p;   // global int4*
typedef __attribute__((address_space(3))) int4v* l_int4p;   // LDS int4*

union AF { half16 v; half8 h[2]; };

__device__ __forceinline__ half8 ldg_h8(const _Float16* p) { return *(const half8*)p; }

__device__ __forceinline__ half16 load_frag2(const _Float16* p0, const _Float16* p1) {
    AF f;
    f.h[0] = *(const half8*)p0;
    f.h[1] = *(const half8*)p1;
    return f.v;
}

__device__ __forceinline__ float8 wmma_f16(half16 a, half16 b, float8 c) {
    return __builtin_amdgcn_wmma_f32_16x16x32_f16(false, a, false, b, (short)0, c, false, false);
}

// ---- async global->LDS staging (gfx1250 ASYNCcnt path), signature per clang-22:
//      (V4i addrspace(1)* src, V4i addrspace(3)* dst, imm offset, imm cpol) ----
#if __has_builtin(__builtin_amdgcn_global_load_async_to_lds_b128)
#define ASYNC_LDS 1
#endif

__device__ __forceinline__ void stage_b128(const _Float16* gsrc, _Float16* ldst) {
#ifdef ASYNC_LDS
    __builtin_amdgcn_global_load_async_to_lds_b128(
        (g_int4p)(__attribute__((address_space(1))) void*)(void*)gsrc,
        (l_int4p)(__attribute__((address_space(3))) void*)ldst, 0, 0);
#else
    *(half8*)ldst = *(const half8*)gsrc;
#endif
}

__device__ __forceinline__ void stage_wait() {
#ifdef ASYNC_LDS
#if __has_builtin(__builtin_amdgcn_s_wait_asynccnt)
    __builtin_amdgcn_s_wait_asynccnt(0);
#else
    asm volatile("s_wait_asynccnt 0" ::: "memory");
#endif
#endif
}

// one-instruction XOR-lane reduction step (stays inside 16-lane halves for m<16)
#define SWZ(x, imm) __int_as_float(__builtin_amdgcn_ds_swizzle(__float_as_int(x), imm))
#define SWZ_X1 0x041f
#define SWZ_X2 0x081f
#define SWZ_X4 0x101f
#define SWZ_X8 0x201f

// ---------- f32 -> f16 convert ----------
__global__ __launch_bounds__(256)
void cvt_f32_f16(const float* __restrict__ in, _Float16* __restrict__ out, size_t n) {
    size_t i = (size_t)blockIdx.x * blockDim.x + threadIdx.x;
    size_t stride = (size_t)gridDim.x * blockDim.x;
    for (; i < n; i += stride) out[i] = (_Float16)in[i];
}

// ---------- GEMM: Y[M,N] = X[M,K] @ W[N,K]^T + bias[N] ----------
// One wave per 16x64 output tile (4 C tiles): each A fragment is reused by 4 WMMAs,
// quartering A-operand traffic vs a 16x16 tile.
// Fragment layouts as loaded here (contiguous k, same order for A and B):
//   A (16x32 f16): lane<16 row=l15 holds K 0..15; lane>=16 holds K 16..31
//   B (32x16 f16): lane<16 col=l15 holds K 0..15; lane>=16 col=l15 holds K 16..31
//   C (16x16 f32): lane<16 col=l15 rows 0..7 in v0-7; lane>=16 rows 8..15
template<bool OUTF32>
__global__ __launch_bounds__(256)
void gemm_xwt(const _Float16* __restrict__ X, const _Float16* __restrict__ W,
              const float* __restrict__ bias, void* __restrict__ Yv,
              int M, int N, int K)
{
    const int lane = threadIdx.x & 31;
    const int wave = threadIdx.x >> 5;
    const int tile = blockIdx.x * 8 + wave;
    const int ntn  = N >> 6;                 // N/64 supertiles
    const int tm   = (tile / ntn) << 4;
    const int tn   = (tile % ntn) << 6;
    const int hl   = lane >> 4;
    const int l15  = lane & 15;

    const _Float16* xrow = X + (size_t)(tm + l15) * K;
    const _Float16* w0   = W + (size_t)(tn + l15) * K;
    const _Float16* w1   = w0 + (size_t)16 * K;
    const _Float16* w2   = w0 + (size_t)32 * K;
    const _Float16* w3   = w0 + (size_t)48 * K;

    float8 c0 = {}, c1 = {}, c2 = {}, c3 = {};
    for (int k = 0; k < K; k += 32) {
        const half16 a = load_frag2(xrow + k + hl * 16, xrow + k + hl * 16 + 8);
        c0 = wmma_f16(a, load_frag2(w0 + k + hl * 16, w0 + k + hl * 16 + 8), c0);
        c1 = wmma_f16(a, load_frag2(w1 + k + hl * 16, w1 + k + hl * 16 + 8), c1);
        c2 = wmma_f16(a, load_frag2(w2 + k + hl * 16, w2 + k + hl * 16 + 8), c2);
        c3 = wmma_f16(a, load_frag2(w3 + k + hl * 16, w3 + k + hl * 16 + 8), c3);
    }

    const float b0 = bias[tn + l15];
    const float b1 = bias[tn + 16 + l15];
    const float b2 = bias[tn + 32 + l15];
    const float b3 = bias[tn + 48 + l15];
    const int rbase = tm + hl * 8;
    if (OUTF32) {
        float* Y = (float*)Yv;
        #pragma unroll
        for (int r = 0; r < 8; ++r) {
            const size_t row = (size_t)(rbase + r) * N + tn + l15;
            Y[row]      = c0[r] + b0;
            Y[row + 16] = c1[r] + b1;
            Y[row + 32] = c2[r] + b2;
            Y[row + 48] = c3[r] + b3;
        }
    } else {
        _Float16* Y = (_Float16*)Yv;
        #pragma unroll
        for (int r = 0; r < 8; ++r) {
            const size_t row = (size_t)(rbase + r) * N + tn + l15;
            Y[row]      = (_Float16)(c0[r] + b0);
            Y[row + 16] = (_Float16)(c1[r] + b1);
            Y[row + 32] = (_Float16)(c2[r] + b2);
            Y[row + 48] = (_Float16)(c3[r] + b3);
        }
    }
}

// ---------- flash attention ----------
// Block = 4 waves (128 threads). Each wave: one 16-row query tile of one (b,h).
// Stream 32-key tiles of K,V through LDS (K via async-to-LDS when available);
// online softmax with ds_swizzle row reductions; O accumulated in f32 WMMA tiles.
__global__ __launch_bounds__(128)
void flash_attn(const _Float16* __restrict__ Qp, const _Float16* __restrict__ Kp,
                const _Float16* __restrict__ Vp, const int* __restrict__ mask,
                _Float16* __restrict__ Oatt)
{
    __shared__ _Float16 Kl[32][64];      // [key][dk]
    __shared__ _Float16 Vt[64][32];      // [dk][key]  (transposed for P@V B-fragments)
    __shared__ _Float16 Pl[4][16][32];   // per-wave P transpose scratch

    const int tid  = threadIdx.x;
    const int lane = tid & 31;
    const int wv   = tid >> 5;
    const int l15  = lane & 15;
    const int hl   = lane >> 4;

    const int qt = blockIdx.x & 63;             // S_/64 query blocks
    const int hh = (blockIdx.x >> 6) & (H_ - 1);
    const int bb = blockIdx.x >> 9;             // / (64*H_)
    const int qb = qt * 64 + wv * 16;

    // Q A-fragments (dk 0..31 and 32..63) stay in registers for the whole loop.
    const _Float16* qrow = Qp + (size_t)(bb * S_ + qb + l15) * D_ + hh * DK_;
    const half16 qa0 = load_frag2(qrow + hl * 16,      qrow + hl * 16 + 8);
    const half16 qa1 = load_frag2(qrow + 32 + hl * 16, qrow + 32 + hl * 16 + 8);

    float8 o0 = {}, o1 = {}, o2 = {}, o3 = {};
    float mrow[8], lrow[8];
    #pragma unroll
    for (int r = 0; r < 8; ++r) { mrow[r] = -3.0e38f; lrow[r] = 0.0f; }

    const int srow = tid >> 2;            // 0..31 (key within tile)
    const int scol = (tid & 3) << 4;      // 0,16,32,48 (dk)
    const float scale = 0.125f;           // 1/sqrt(64)

    for (int kb = 0; kb < S_; kb += 32) {
        // ---- stage K (async-to-LDS) and transposed V tiles (128 threads) ----
        {
            const _Float16* ksrc = Kp + (size_t)(bb * S_ + kb + srow) * D_ + hh * DK_ + scol;
            stage_b128(ksrc,     &Kl[srow][scol]);
            stage_b128(ksrc + 8, &Kl[srow][scol + 8]);

            const _Float16* vsrc = Vp + (size_t)(bb * S_ + kb + srow) * D_ + hh * DK_ + scol;
            half8 v0 = ldg_h8(vsrc), v1 = ldg_h8(vsrc + 8);
            #pragma unroll
            for (int j = 0; j < 8; ++j) {
                Vt[scol + j][srow]     = v0[j];
                Vt[scol + 8 + j][srow] = v1[j];
            }
            if (kb + 32 < S_) {
                __builtin_prefetch(ksrc + (size_t)32 * D_, 0, 0);  // next tiles -> global_prefetch
                __builtin_prefetch(vsrc + (size_t)32 * D_, 0, 0);
            }
        }
        stage_wait();
        __syncthreads();

        // ---- scores: S = Q @ K^T, two 16x16 key tiles, K-dim 64 = 2 WMMAs each ----
        float8 s0 = {}, s1 = {};
        s0 = wmma_f16(qa0, load_frag2(&Kl[l15][hl * 16],      &Kl[l15][hl * 16 + 8]),      s0);
        s0 = wmma_f16(qa1, load_frag2(&Kl[l15][32 + hl * 16], &Kl[l15][32 + hl * 16 + 8]), s0);
        s1 = wmma_f16(qa0, load_frag2(&Kl[16 + l15][hl * 16],      &Kl[16 + l15][hl * 16 + 8]),      s1);
        s1 = wmma_f16(qa1, load_frag2(&Kl[16 + l15][32 + hl * 16], &Kl[16 + l15][32 + hl * 16 + 8]), s1);

        // ---- scale + mask (mask is [B,1,1,S]; this lane's key column) ----
        const int msk0 = mask[bb * S_ + kb + l15];
        const int msk1 = mask[bb * S_ + kb + 16 + l15];
        #pragma unroll
        for (int r = 0; r < 8; ++r) {
            float a = s0[r] * scale; if (msk0 == 0) a = -1.0e9f; s0[r] = a;
            float b = s1[r] * scale; if (msk1 == 0) b = -1.0e9f; s1[r] = b;
        }

        // ---- online softmax (row stats live per 16-lane half; ds_swizzle XOR steps) ----
        float alpha[8];
        #pragma unroll
        for (int r = 0; r < 8; ++r) {
            float mx = fmaxf(s0[r], s1[r]);
            mx = fmaxf(mx, SWZ(mx, SWZ_X1));
            mx = fmaxf(mx, SWZ(mx, SWZ_X2));
            mx = fmaxf(mx, SWZ(mx, SWZ_X4));
            mx = fmaxf(mx, SWZ(mx, SWZ_X8));
            const float mnew = fmaxf(mrow[r], mx);
            const float p0 = __expf(s0[r] - mnew);
            const float p1 = __expf(s1[r] - mnew);
            s0[r] = p0; s1[r] = p1;
            float rs = p0 + p1;
            rs += SWZ(rs, SWZ_X1);
            rs += SWZ(rs, SWZ_X2);
            rs += SWZ(rs, SWZ_X4);
            rs += SWZ(rs, SWZ_X8);
            alpha[r] = __expf(mrow[r] - mnew);
            lrow[r] = alpha[r] * lrow[r] + rs;
            mrow[r] = mnew;
        }
        #pragma unroll
        for (int r = 0; r < 8; ++r) {
            o0[r] *= alpha[r]; o1[r] *= alpha[r]; o2[r] *= alpha[r]; o3[r] *= alpha[r];
        }

        // ---- transpose P (C layout -> A layout) through per-wave LDS scratch ----
        const int prow = hl * 8;
        #pragma unroll
        for (int r = 0; r < 8; ++r) {
            Pl[wv][prow + r][l15]      = (_Float16)s0[r];
            Pl[wv][prow + r][16 + l15] = (_Float16)s1[r];
        }
        asm volatile("s_wait_dscnt 0" ::: "memory");
        const half16 pa = load_frag2(&Pl[wv][l15][hl * 16], &Pl[wv][l15][hl * 16 + 8]);

        // ---- O += P @ V  (4 dk groups, K-dim = 32 keys) ----
        o0 = wmma_f16(pa, load_frag2(&Vt[l15][hl * 16],      &Vt[l15][hl * 16 + 8]),      o0);
        o1 = wmma_f16(pa, load_frag2(&Vt[16 + l15][hl * 16], &Vt[16 + l15][hl * 16 + 8]), o1);
        o2 = wmma_f16(pa, load_frag2(&Vt[32 + l15][hl * 16], &Vt[32 + l15][hl * 16 + 8]), o2);
        o3 = wmma_f16(pa, load_frag2(&Vt[48 + l15][hl * 16], &Vt[48 + l15][hl * 16 + 8]), o3);

        __syncthreads();
    }

    // ---- normalize and store attention output (f16, [B*S, D] layout) ----
    #pragma unroll
    for (int r = 0; r < 8; ++r) {
        const float inv = 1.0f / lrow[r];
        const size_t base = (size_t)(bb * S_ + qb + hl * 8 + r) * D_ + hh * DK_ + l15;
        Oatt[base]      = (_Float16)(o0[r] * inv);
        Oatt[base + 16] = (_Float16)(o1[r] * inv);
        Oatt[base + 32] = (_Float16)(o2[r] * inv);
        Oatt[base + 48] = (_Float16)(o3[r] * inv);
    }
}

// ---------- host-side orchestration ----------
extern "C" void kernel_launch(void* const* d_in, const int* in_sizes, int n_in,
                              void* d_out, int out_size, void* d_ws, size_t ws_size,
                              hipStream_t stream) {
    (void)in_sizes; (void)n_in; (void)out_size; (void)ws_size;
    const float* q    = (const float*)d_in[0];
    const float* k    = (const float*)d_in[1];
    const float* v    = (const float*)d_in[2];
    const int*   mask = (const int*)  d_in[3];
    const float* Wq = (const float*)d_in[4];  const float* bq = (const float*)d_in[5];
    const float* Wk = (const float*)d_in[6];  const float* bk = (const float*)d_in[7];
    const float* Wv = (const float*)d_in[8];  const float* bv = (const float*)d_in[9];
    const float* Wo = (const float*)d_in[10]; const float* bo = (const float*)d_in[11];

    const size_t NX = (size_t)B_ * S_ * D_;   // 4,194,304 elements
    const size_t NW = (size_t)D_ * D_;        //   262,144 elements

    char* ws = (char*)d_ws;
    _Float16* Xq  = (_Float16*)ws; ws += NX * sizeof(_Float16);
    _Float16* Xk  = (_Float16*)ws; ws += NX * sizeof(_Float16);
    _Float16* Xv  = (_Float16*)ws; ws += NX * sizeof(_Float16);
    _Float16* Wqf = (_Float16*)ws; ws += NW * sizeof(_Float16);
    _Float16* Wkf = (_Float16*)ws; ws += NW * sizeof(_Float16);
    _Float16* Wvf = (_Float16*)ws; ws += NW * sizeof(_Float16);
    _Float16* Wof = (_Float16*)ws; ws += NW * sizeof(_Float16);
    _Float16* Qp  = (_Float16*)ws; ws += NX * sizeof(_Float16);
    _Float16* Kp  = (_Float16*)ws; ws += NX * sizeof(_Float16);
    _Float16* Vp  = (_Float16*)ws; ws += NX * sizeof(_Float16);
    _Float16* Oa  = (_Float16*)ws; ws += NX * sizeof(_Float16);

    // 1) precision conversion
    cvt_f32_f16<<<2048, 256, 0, stream>>>(q,  Xq,  NX);
    cvt_f32_f16<<<2048, 256, 0, stream>>>(k,  Xk,  NX);
    cvt_f32_f16<<<2048, 256, 0, stream>>>(v,  Xv,  NX);
    cvt_f32_f16<<<256,  256, 0, stream>>>(Wq, Wqf, NW);
    cvt_f32_f16<<<256,  256, 0, stream>>>(Wk, Wkf, NW);
    cvt_f32_f16<<<256,  256, 0, stream>>>(Wv, Wvf, NW);
    cvt_f32_f16<<<256,  256, 0, stream>>>(Wo, Wof, NW);

    // 2) Q/K/V projections: [8192,512] = X @ W^T + b
    //    (M/16)*(N/64) = 4096 wave-tiles / 8 waves per block = 512 blocks
    const int M = B_ * S_, N = D_, K = D_;
    const int gemm_blocks = (M / 16) * (N / 64) / 8;   // 512
    gemm_xwt<false><<<gemm_blocks, 256, 0, stream>>>(Xq, Wqf, bq, Qp, M, N, K);
    gemm_xwt<false><<<gemm_blocks, 256, 0, stream>>>(Xk, Wkf, bk, Kp, M, N, K);
    gemm_xwt<false><<<gemm_blocks, 256, 0, stream>>>(Xv, Wvf, bv, Vp, M, N, K);

    // 3) flash attention: B*H*(S/64) = 1024 blocks, 4 waves each
    flash_attn<<<B_ * H_ * (S_ / 64), 128, 0, stream>>>(Qp, Kp, Vp, mask, Oa);

    // 4) output projection, fp32 result
    gemm_xwt<true><<<gemm_blocks, 256, 0, stream>>>(Oa, Wof, bo, (float*)d_out, M, N, K);
}
